// NeuralHMM_8186207666608
// MI455X (gfx1250) — matrix operand order repaired and checked by reading, not compiled, not run on hardware
//
#include <hip/hip_runtime.h>
#include <hip/hip_bf16.h>

// B=16, T=1024, D_EMB=1024, H=64
// M = B*T = 16384, K = 1024, N = H*H = 4096

typedef __attribute__((ext_vector_type(16))) __bf16 v16bf;
typedef __attribute__((ext_vector_type(8)))  __bf16 v8bf;
typedef __attribute__((ext_vector_type(8)))  float  v8f;
typedef int vint4 __attribute__((vector_size(16)));   // matches builtin param

#define M_TOT 16384
#define K_TOT 1024
#define N_TOT 4096
#define HB    16        // batch
#define HT    1024      // time
#define HH    64        // hidden states

#define AS1 __attribute__((address_space(1)))
#define AS3 __attribute__((address_space(3)))

// ---------------------------------------------------------------------------
// Kernel 0: row softmax of the static 64x64 transition matrix
// ---------------------------------------------------------------------------
__global__ void row_softmax64(const float* __restrict__ u, float* __restrict__ o) {
    const int i = blockIdx.x;      // row
    const int j = threadIdx.x;     // col, 64 threads (2 waves)
    __shared__ float sh[64];
    __shared__ float redmax;
    __shared__ float redsum;
    float v = u[i * 64 + j];
    sh[j] = v;
    __syncthreads();
    if (j < 32) {
        float m = fmaxf(sh[j], sh[j + 32]);
        for (int off = 16; off; off >>= 1) m = fmaxf(m, __shfl_xor(m, off, 32));
        if (j == 0) redmax = m;
    }
    __syncthreads();
    float e = __expf(v - redmax);
    sh[j] = e;
    __syncthreads();
    if (j < 32) {
        float s = sh[j] + sh[j + 32];
        for (int off = 16; off; off >>= 1) s += __shfl_xor(s, off, 32);
        if (j == 0) redsum = s;
    }
    __syncthreads();
    o[i * 64 + j] = e / redsum;
}

// ---------------------------------------------------------------------------
// Kernel 1a: fp32 -> bf16 convert (emb), exact-sized launch
// ---------------------------------------------------------------------------
__global__ void cvt_bf16(const float* __restrict__ src, __bf16* __restrict__ dst) {
    const int i = blockIdx.x * 256 + threadIdx.x;
    dst[i] = (__bf16)src[i];
}

// ---------------------------------------------------------------------------
// Kernel 1b: fp32 W[K][N] -> bf16 W^T[N][K] (one-time 16MB transpose)
// ---------------------------------------------------------------------------
__global__ void cvt_transpose_bf16(const float* __restrict__ W, __bf16* __restrict__ WT) {
    const int i = blockIdx.x * 256 + threadIdx.x;     // i over K_TOT*N_TOT
    const int k = i >> 12;                            // / 4096
    const int n = i & 4095;
    WT[(size_t)n * K_TOT + k] = (__bf16)W[i];
}

// ---------------------------------------------------------------------------
// Kernel 2: WMMA GEMM (bf16 -> f32) + bias + fused row-softmax over the H=64
// column group + mix with static transition row.  Writes linear mixture
//   mix[b,t,i,j] = 0.5*trans_sm[i,j] + 0.5*softmax_j(emb@W + b)
//
// 256 threads = 8 waves, tile 64(M) x 64(N), K stepped by 64.
// LDS staging uses GLOBAL_LOAD_ASYNC_TO_LDS_B128 (ASYNCcnt) with double
// buffering: stage k+1 while computing k.  4 WMMAs per wave per stage.
// ---------------------------------------------------------------------------
__global__ __launch_bounds__(256) void gemm_softmax_mix(
    const __bf16* __restrict__ A,        // [M_TOT][K_TOT]
    const __bf16* __restrict__ WT,       // [N_TOT][K_TOT]  (W transposed)
    const float*  __restrict__ bias,     // [N_TOT]
    const float*  __restrict__ trans_sm, // [64][64]
    float*        __restrict__ mix)      // [M_TOT][N_TOT]
{
    __shared__ __align__(16) __bf16 As[2][64 * 64];  // [buf][row][k] pitch 64
    __shared__ __align__(16) __bf16 Bs[2][64 * 64];  // [buf][n]  [k] pitch 64
    __shared__ float ftile[64 * 66];                 // padded f32 result tile
    __shared__ float rowm[64];
    __shared__ float rowinv[64];
    __shared__ float trow[64];

    const int tid   = threadIdx.x;
    const int tileN = blockIdx.x;          // 0..63   -> h_row index
    const int tileM = blockIdx.y;          // 0..255
    const int m0 = tileM * 64;
    const int n0 = tileN * 64;

    if (tid < 64) trow[tid] = trans_sm[tileN * 64 + tid];

    const int lane = tid & 31;
    const int wid  = tid >> 5;
    const int wr   = wid & 3;              // wave row-tile  (0..3)
    const int wc   = wid >> 2;             // wave col-tile  (0..1)
    const int mfr  = lane & 15;            // M row (A) / N col (B) within frag
    const int half = lane >> 4;            // lane-half selects K window

    v8f acc0 = {};
    v8f acc1 = {};

    // staging: each thread async-copies 2x16B of A and 2x16B of WT per stage
    const int srow  = tid >> 2;            // 0..63
    const int scol8 = (tid & 3) * 8;       // bf16 elems: 0,8,16,24 (+32 for 2nd)
    const __bf16* gArow = A  + (size_t)(m0 + srow) * K_TOT + scol8;
    const __bf16* gBrow = WT + (size_t)(n0 + srow) * K_TOT + scol8;

    auto stage = [&](int buf, int k0) {
        const __bf16* ga = gArow + k0;
        const __bf16* gb = gBrow + k0;
        __bf16* la = &As[buf][srow * 64 + scol8];
        __bf16* lb = &Bs[buf][srow * 64 + scol8];
        __builtin_amdgcn_global_load_async_to_lds_b128(
            (AS1 vint4*)ga,        (AS3 vint4*)la,        0, 0);
        __builtin_amdgcn_global_load_async_to_lds_b128(
            (AS1 vint4*)(ga + 32), (AS3 vint4*)(la + 32), 0, 0);
        __builtin_amdgcn_global_load_async_to_lds_b128(
            (AS1 vint4*)gb,        (AS3 vint4*)lb,        0, 0);
        __builtin_amdgcn_global_load_async_to_lds_b128(
            (AS1 vint4*)(gb + 32), (AS3 vint4*)(lb + 32), 0, 0);
    };

    stage(0, 0);

    for (int k0 = 0, it = 0; k0 < K_TOT; k0 += 64, ++it) {
        const int buf = it & 1;
        // this wave's stage(it) async loads complete, then block-wide barrier
        asm volatile("s_wait_asynccnt 0x0" ::: "memory");
        __syncthreads();
        if (k0 + 64 < K_TOT) stage(buf ^ 1, k0 + 64);  // overlap with compute

        #pragma unroll
        for (int ks = 0; ks < 64; ks += 32) {
            // A fragment: lane(0-15) row M=mfr, K = half*16 + (0..15)
            const v8bf* ap = (const v8bf*)&As[buf][(wr * 16 + mfr) * 64 + ks + half * 16];
            union { v16bf v; v8bf h[2]; } af;
            af.h[0] = ap[0];
            af.h[1] = ap[1];   // +8 bf16 elements

            // B fragments: lane(0-15)=N col, K = half*16 + (0..15)
            const v16bf bf0 = *(const v16bf*)&Bs[buf][(wc * 32 +      mfr) * 64 + ks + half * 16];
            const v16bf bf1 = *(const v16bf*)&Bs[buf][(wc * 32 + 16 + mfr) * 64 + ks + half * 16];

            acc0 = __builtin_amdgcn_wmma_f32_16x16x32_bf16(
                false, af.v, false, bf0, (short)0, acc0, false, false);
            acc1 = __builtin_amdgcn_wmma_f32_16x16x32_bf16(
                false, af.v, false, bf1, (short)0, acc1, false, false);
        }
        __syncthreads();   // compute(it) done before buf is overwritten (it+2)
    }

    // C/D layout: VGPR v -> M = v + (lane>=16 ? 8 : 0), N = lane&15
    const int rbase = wr * 16 + half * 8;
    #pragma unroll
    for (int v = 0; v < 8; ++v) {
        const int r  = rbase + v;
        const int c0 = wc * 32 + mfr;
        const int c1 = wc * 32 + 16 + mfr;
        ftile[r * 66 + c0] = acc0[v] + bias[n0 + c0];
        ftile[r * 66 + c1] = acc1[v] + bias[n0 + c1];
    }
    __syncthreads();

    if (tid < 64) {
        float m = -3.402823466e38f;
        for (int c = 0; c < 64; ++c) m = fmaxf(m, ftile[tid * 66 + c]);
        float s = 0.f;
        for (int c = 0; c < 64; ++c) s += __expf(ftile[tid * 66 + c] - m);
        rowm[tid]   = m;
        rowinv[tid] = 1.0f / s;
    }
    __syncthreads();

    const int r  = tid >> 2;
    const int cb = (tid & 3) * 16;
    float* outp = mix + (size_t)(m0 + r) * N_TOT + n0 + cb;
    const float rm = rowm[r];
    const float ri = rowinv[r];
    #pragma unroll
    for (int c16 = 0; c16 < 16; ++c16) {
        const int c = cb + c16;
        const float p = __expf(ftile[r * 66 + c] - rm) * ri;
        outp[c16] = 0.5f * trow[c] + 0.5f * p;
    }
}

// ---------------------------------------------------------------------------
// Kernel 3: HMM forward scan in log space using linear mixture matrices.
//   h_new[j] = mh + log( sum_i exp(h_i - mh) * mix[b,t,i,j] )
// One block per batch element, 256 threads: 4 segments x 64 states.
// ---------------------------------------------------------------------------
__global__ __launch_bounds__(256) void hmm_scan(
    const float* __restrict__ mix,          // [B][T][H][H]
    const float* __restrict__ state_priors, // [H]
    float*       __restrict__ out)          // [B][T][H]
{
    const int b   = blockIdx.x;
    const int tid = threadIdx.x;
    const int j   = tid & 63;
    const int seg = tid >> 6;               // 0..3

    __shared__ float h[64];
    __shared__ float p[64];
    __shared__ float part[4][64];
    __shared__ float redmax;

    if (tid < 64) {
        const float v = state_priors[tid];
        h[tid] = v;
        out[(size_t)b * HT * HH + tid] = v;   // t = 0 row
    }
    __syncthreads();

    const float* Mb = mix + (size_t)b * HT * (HH * HH);

    for (int t = 1; t < HT; ++t) {
        // block max of h (64 values): fold + wave32 shuffle reduce
        if (tid < 32) {
            float m = fmaxf(h[tid], h[tid + 32]);
            for (int off = 16; off; off >>= 1) m = fmaxf(m, __shfl_xor(m, off, 32));
            if (tid == 0) redmax = m;
        }
        __syncthreads();
        const float mh = redmax;
        if (tid < 64) p[tid] = __expf(h[tid] - mh);

        // prefetch next timestep's 16KB mixture matrix (global_prefetch_b8)
        if (t + 1 < HT)
            __builtin_prefetch(Mb + (size_t)(t + 1) * (HH * HH) + tid * 16, 0, 1);
        __syncthreads();

        // partial dot: segment of 16 i-values, column j
        const float* Mt = Mb + (size_t)t * (HH * HH) + seg * 16 * HH + j;
        float s = 0.f;
        #pragma unroll
        for (int i = 0; i < 16; ++i) s = fmaf(p[seg * 16 + i], Mt[(size_t)i * HH], s);
        part[seg][j] = s;
        __syncthreads();

        if (tid < 64) {
            const float sum = part[0][j] + part[1][j] + part[2][j] + part[3][j];
            const float hn  = mh + __logf(sum);
            h[j] = hn;
            out[((size_t)b * HT + t) * HH + j] = hn;
        }
        __syncthreads();
    }
}

// ---------------------------------------------------------------------------
// Launch
// ---------------------------------------------------------------------------
extern "C" void kernel_launch(void* const* d_in, const int* in_sizes, int n_in,
                              void* d_out, int out_size, void* d_ws, size_t ws_size,
                              hipStream_t stream) {
    const float* emb    = (const float*)d_in[0];  // [16,1024,1024]
    const float* W      = (const float*)d_in[1];  // [1024,4096]
    const float* bias   = (const float*)d_in[2];  // [4096]
    const float* unorm  = (const float*)d_in[3];  // [64,64]
    const float* priors = (const float*)d_in[4];  // [64]
    float* out = (float*)d_out;                   // [16,1024,64]

    char* ws = (char*)d_ws;
    __bf16* Abf  = (__bf16*)(ws);                       // 33,554,432 B
    __bf16* WTbf = (__bf16*)(ws + 33554432u);           //  8,388,608 B
    float*  tsm  = (float*) (ws + 41943040u);           //     16,384 B
    float*  mixb = (float*) (ws + 41959424u);           // 268,435,456 B

    row_softmax64<<<64, 64, 0, stream>>>(unorm, tsm);
    cvt_bf16<<<(M_TOT * K_TOT) / 256, 256, 0, stream>>>(emb, Abf);
    cvt_transpose_bf16<<<(K_TOT * N_TOT) / 256, 256, 0, stream>>>(W, WTbf);

    dim3 g2(N_TOT / 64, M_TOT / 64);   // (64, 256)
    gemm_softmax_mix<<<g2, 256, 0, stream>>>(Abf, WTbf, bias, tsm, mixb);

    hmm_scan<<<HB, 256, 0, stream>>>(mixb, priors, out);
}
